// Class_Attention_46583215292893
// MI455X (gfx1250) — compile-verified, matrix operand order checked by reading, not run on hardware
//
#include <hip/hip_runtime.h>
#include <hip/hip_bf16.h>
#include <stdint.h>

// ---------------------------------------------------------------------------
// Problem constants (B,N,C,H from the reference)
// ---------------------------------------------------------------------------
#define BB     16
#define NN     577
#define NPAD   608            // 38*16 and 19*32: K-loops and 32-row blocks tile exactly
#define NT     38             // NPAD / 16
#define CC     768
#define HH     12
#define HDIM   64
#define ROWS   (BB * NN)      // 9232
#define BHD    (BB * HH)      // 192 (b,h) attention matrices
#define SCALE_F 0.125f        // 64^-0.5
#define H_HOP_F 4.0f

// ---------------------------------------------------------------------------
// WMMA helpers: v_wmma_f32_16x16x32_bf16, wave32 operand layouts per ISA 7.12.2
//   A: lane L holds row (L&15); element i -> K = (i<8 ? i : i+8) + (L>>4)*8
//      => two contiguous 16-byte loads per lane at k0 + half*8 and k0+16+half*8
//   B: lane L holds column (L&15), same K packing (B operands stored so a
//      column is contiguous along K)
//   C/D: lane L element r -> (M = r + (L>>4)*8, N = L&15)
// ---------------------------------------------------------------------------
typedef __attribute__((ext_vector_type(16))) __bf16 v16bf;
typedef __attribute__((ext_vector_type(8)))  float  v8f;

union ABu { v16bf v; uint4 u[2]; };

__device__ __forceinline__ v16bf load_k32(const __bf16* rowptr, int half) {
    ABu t;
    t.u[0] = *(const uint4*)(rowptr + half * 8);
    t.u[1] = *(const uint4*)(rowptr + 16 + half * 8);
    return t.v;
}

__device__ __forceinline__ v8f wmma_bf16(v16bf a, v16bf b, v8f c) {
    return __builtin_amdgcn_wmma_f32_16x16x32_bf16(
        /*neg_a=*/false, a, /*neg_b=*/false, b,
        /*c_mod=*/(short)0, c, /*reuse_a=*/false, /*reuse_b=*/false);
}

// ---------------------------------------------------------------------------
// Kernel 0: fp32 -> bf16 convert (x and the four weight matrices)
// ---------------------------------------------------------------------------
__global__ void cvt_bf16(const float* __restrict__ s, __bf16* __restrict__ d, int n) {
    int i = blockIdx.x * blockDim.x + threadIdx.x;
    if (i < n) d[i] = (__bf16)s[i];
}

// ---------------------------------------------------------------------------
// Kernel 1: fused QKV projection, 32x32 per wave (2x2 WMMA tiles).
//   Q,K stored [bh][n_pad][64] bf16 (pad rows zeroed); SCALE folded into Q.
//   V stored transposed [bh][64][n_pad].
// grid = (19 row blocks, B, 72 = {q,k,v} x 24 col blocks), block = 32
// ---------------------------------------------------------------------------
__global__ void __launch_bounds__(32)
qkv_gemm(const __bf16* __restrict__ xb,  const __bf16* __restrict__ wq,
         const __bf16* __restrict__ wk,  const __bf16* __restrict__ wv,
         __bf16* __restrict__ qb, __bf16* __restrict__ kb,
         __bf16* __restrict__ vtb)
{
    const int lane = threadIdx.x & 31;
    const int l16  = lane & 15, half = lane >> 4;
    const int rb = blockIdx.x;            // 32-row block within padded batch (0..18)
    const int b  = blockIdx.y;
    const int which = blockIdx.z / 24;    // 0=q 1=k 2=v
    const int cb    = blockIdx.z % 24;    // 32-col block

    const __bf16* w = (which == 0) ? wq : (which == 1) ? wk : wv;

    int n0 = rb * 32 + l16;
    int n1 = n0 + 16;
    const __bf16* arow0 = xb + (size_t)(b * NN + (n0 < NN ? n0 : NN - 1)) * CC;
    const __bf16* arow1 = xb + (size_t)(b * NN + (n1 < NN ? n1 : NN - 1)) * CC;
    const __bf16* brow0 = w + (size_t)(cb * 32 + l16) * CC;
    const __bf16* brow1 = brow0 + (size_t)16 * CC;
    __builtin_prefetch(arow0, 0, 3);      // global_prefetch_b8
    __builtin_prefetch(arow1, 0, 3);
    __builtin_prefetch(brow0, 0, 3);
    __builtin_prefetch(brow1, 0, 3);

    v8f c00 = {}, c01 = {}, c10 = {}, c11 = {};
    for (int k0 = 0; k0 < CC; k0 += 32) {
        v16bf a0 = load_k32(arow0 + k0, half);
        v16bf a1 = load_k32(arow1 + k0, half);
        v16bf b0 = load_k32(brow0 + k0, half);
        v16bf b1 = load_k32(brow1 + k0, half);
        c00 = wmma_bf16(a0, b0, c00);
        c01 = wmma_bf16(a0, b1, c01);
        c10 = wmma_bf16(a1, b0, c10);
        c11 = wmma_bf16(a1, b1, c11);
    }

    const float sc = (which == 0) ? SCALE_F : 1.0f;
#pragma unroll
    for (int i = 0; i < 2; ++i) {
        v8f* row = (i == 0) ? &c00 : &c10;
#pragma unroll
        for (int j = 0; j < 2; ++j) {
            v8f acc = (i == 0) ? (j == 0 ? c00 : c01) : (j == 0 ? c10 : c11);
            (void)row;
#pragma unroll
            for (int r = 0; r < 8; ++r) {
                int M = r + half * 8;
                int n = rb * 32 + i * 16 + M;
                int co = cb * 32 + j * 16 + l16;
                int h = co >> 6, dd = co & 63;
                float v = (n < NN) ? acc[r] * sc : 0.0f;  // zero padding rows
                __bf16 bv = (__bf16)v;
                size_t bh = (size_t)(b * HH + h);
                if (which == 2)
                    vtb[(bh * HDIM + dd) * NPAD + n] = bv;
                else {
                    __bf16* dst = (which == 0) ? qb : kb;
                    dst[(bh * NPAD + n) * HDIM + dd] = bv;
                }
            }
        }
    }
}

// ---------------------------------------------------------------------------
// Kernel 2: logits (Q K^T) + row softmax -> bf16 attn [bh][n_pad][n_pad].
//   One workgroup (8 waves) owns a 16-row x 608-col block; WMMA tiles land in
//   LDS, then 16 threads/row do max/exp/sum with width-16 shuffles (wave32).
//   Pad columns forced to -1e30 pre-softmax and exact 0 post; pad rows -> 0.
// grid = (38, H, B), block = 256
// ---------------------------------------------------------------------------
__global__ void __launch_bounds__(256)
attn_softmax(const __bf16* __restrict__ qb, const __bf16* __restrict__ kb,
             __bf16* __restrict__ attn)
{
    __shared__ float lg[16][NPAD + 8];
    const int tid  = threadIdx.x;
    const int wave = tid >> 5, lane = tid & 31;
    const int l16  = lane & 15, half = lane >> 4;
    const int mt   = blockIdx.x;
    const size_t bh = (size_t)blockIdx.z * HH + blockIdx.y;

    const __bf16* arow = qb + (bh * NPAD + mt * 16 + l16) * HDIM;
    for (int ct = wave; ct < NT; ct += 8) {
        const __bf16* brow = kb + (bh * NPAD + ct * 16 + l16) * HDIM;
        v8f acc = {};
        acc = wmma_bf16(load_k32(arow, half),      load_k32(brow, half),      acc);
        acc = wmma_bf16(load_k32(arow + 32, half), load_k32(brow + 32, half), acc);
        int col = ct * 16 + l16;
#pragma unroll
        for (int r = 0; r < 8; ++r) {
            int M = r + half * 8;
            lg[M][col] = (col < NN) ? acc[r] : -1e30f;
        }
    }
    __syncthreads();

    const int r = tid >> 4, t = tid & 15;       // 16 threads per row
    float mx = -1e30f;
    for (int c = t; c < NPAD; c += 16) mx = fmaxf(mx, lg[r][c]);
#pragma unroll
    for (int s = 1; s < 16; s <<= 1) mx = fmaxf(mx, __shfl_xor(mx, s, 16));
    float sum = 0.0f;
    for (int c = t; c < NPAD; c += 16) {
        float e = __expf(lg[r][c] - mx);
        lg[r][c] = e;
        sum += e;
    }
#pragma unroll
    for (int s = 1; s < 16; s <<= 1) sum += __shfl_xor(sum, s, 16);
    float inv = 1.0f / sum;

    int n = mt * 16 + r;
    bool vr = n < NN;
    __bf16* orow = attn + (bh * NPAD + n) * NPAD;
    for (int c = t; c < NPAD; c += 16)
        orow[c] = (vr && c < NN) ? (__bf16)(lg[r][c] * inv) : (__bf16)0.0f;
}

// ---------------------------------------------------------------------------
// Kernel 3: Y = attn @ V, 32 rows x 32 head-cols per wave.
//   Stored transposed yT[bh][64][n_pad].  608 = 19*32 rows tile exactly.
// grid = (19 row blocks, 2 col blocks, 192 bh), block = 32
// ---------------------------------------------------------------------------
__global__ void __launch_bounds__(32)
av_gemm(const __bf16* __restrict__ attn, const __bf16* __restrict__ vtb,
        __bf16* __restrict__ ytb)
{
    const int lane = threadIdx.x & 31;
    const int l16  = lane & 15, half = lane >> 4;
    const size_t bh = blockIdx.z;
    const int rb = blockIdx.x, db = blockIdx.y;

    const __bf16* arow0 = attn + (bh * NPAD + rb * 32 + l16) * NPAD;
    const __bf16* arow1 = arow0 + (size_t)16 * NPAD;
    const __bf16* brow0 = vtb + (bh * HDIM + db * 32 + l16) * NPAD;
    const __bf16* brow1 = brow0 + (size_t)16 * NPAD;
    __builtin_prefetch(arow0, 0, 3);
    __builtin_prefetch(arow1, 0, 3);
    __builtin_prefetch(brow0, 0, 3);
    __builtin_prefetch(brow1, 0, 3);

    v8f c00 = {}, c01 = {}, c10 = {}, c11 = {};
    for (int k0 = 0; k0 < NPAD; k0 += 32) {
        v16bf a0 = load_k32(arow0 + k0, half);
        v16bf a1 = load_k32(arow1 + k0, half);
        v16bf b0 = load_k32(brow0 + k0, half);
        v16bf b1 = load_k32(brow1 + k0, half);
        c00 = wmma_bf16(a0, b0, c00);
        c01 = wmma_bf16(a0, b1, c01);
        c10 = wmma_bf16(a1, b0, c10);
        c11 = wmma_bf16(a1, b1, c11);
    }

#pragma unroll
    for (int i = 0; i < 2; ++i)
#pragma unroll
    for (int j = 0; j < 2; ++j) {
        v8f acc = (i == 0) ? (j == 0 ? c00 : c01) : (j == 0 ? c10 : c11);
        int d = db * 32 + j * 16 + l16;
#pragma unroll
        for (int r = 0; r < 8; ++r) {
            int M = r + half * 8;
            int n = rb * 32 + i * 16 + M;
            ytb[(bh * HDIM + d) * NPAD + n] = (__bf16)acc[r];
        }
    }
}

// ---------------------------------------------------------------------------
// Kernel 4: Z = attn @ Y ; out_head = (1 + beta*(2-HOP))*Y + beta*(HOP-1)*Z
//   (exact restructuring of the reference's (attn@attn)@v term)
//   out_head stored bf16 [b*N][C] for the final projection GEMM.
// grid = (19, 2, 192), block = 32
// ---------------------------------------------------------------------------
__global__ void __launch_bounds__(32)
az_combine(const __bf16* __restrict__ attn, const __bf16* __restrict__ ytb,
           const float* __restrict__ lamb, __bf16* __restrict__ ohb)
{
    const int lane = threadIdx.x & 31;
    const int l16  = lane & 15, half = lane >> 4;
    const int bhz = blockIdx.z;
    const int b = bhz / HH, h = bhz % HH;
    const size_t bh = (size_t)bhz;
    const int rb = blockIdx.x, db = blockIdx.y;

    const __bf16* arow0 = attn + (bh * NPAD + rb * 32 + l16) * NPAD;
    const __bf16* arow1 = arow0 + (size_t)16 * NPAD;
    const __bf16* brow0 = ytb + (bh * HDIM + db * 32 + l16) * NPAD;
    const __bf16* brow1 = brow0 + (size_t)16 * NPAD;
    __builtin_prefetch(arow0, 0, 3);
    __builtin_prefetch(arow1, 0, 3);
    __builtin_prefetch(brow0, 0, 3);
    __builtin_prefetch(brow1, 0, 3);

    v8f c00 = {}, c01 = {}, c10 = {}, c11 = {};
    for (int k0 = 0; k0 < NPAD; k0 += 32) {
        v16bf a0 = load_k32(arow0 + k0, half);
        v16bf a1 = load_k32(arow1 + k0, half);
        v16bf b0 = load_k32(brow0 + k0, half);
        v16bf b1 = load_k32(brow1 + k0, half);
        c00 = wmma_bf16(a0, b0, c00);
        c01 = wmma_bf16(a0, b1, c01);
        c10 = wmma_bf16(a1, b0, c10);
        c11 = wmma_bf16(a1, b1, c11);
    }

    float beta = lamb[h];
    float k1 = 1.0f + beta * (2.0f - H_HOP_F);
    float k2 = beta * (H_HOP_F - 1.0f);
#pragma unroll
    for (int i = 0; i < 2; ++i)
#pragma unroll
    for (int j = 0; j < 2; ++j) {
        v8f acc = (i == 0) ? (j == 0 ? c00 : c01) : (j == 0 ? c10 : c11);
        int d = db * 32 + j * 16 + l16;
#pragma unroll
        for (int r = 0; r < 8; ++r) {
            int M = r + half * 8;
            int n = rb * 32 + i * 16 + M;
            if (n < NN) {
                float y = (float)ytb[(bh * HDIM + d) * NPAD + n];
                ohb[((size_t)b * NN + n) * CC + h * HDIM + d] =
                    (__bf16)(k1 * y + k2 * acc[r]);
            }
        }
    }
}

// ---------------------------------------------------------------------------
// Kernel 5: out = out_head @ Wp^T + bp (fp32 result), 32x32 per wave.
// grid = (289 row blocks, 24 col blocks), block = 32
// ---------------------------------------------------------------------------
__global__ void __launch_bounds__(32)
out_proj(const __bf16* __restrict__ ohb, const __bf16* __restrict__ wp,
         const float* __restrict__ bp, float* __restrict__ out)
{
    const int lane = threadIdx.x & 31;
    const int l16  = lane & 15, half = lane >> 4;
    const int rb = blockIdx.x, cb = blockIdx.y;

    int m0 = rb * 32 + l16;
    int m1 = m0 + 16;
    const __bf16* arow0 = ohb + (size_t)(m0 < ROWS ? m0 : ROWS - 1) * CC;
    const __bf16* arow1 = ohb + (size_t)(m1 < ROWS ? m1 : ROWS - 1) * CC;
    const __bf16* brow0 = wp + (size_t)(cb * 32 + l16) * CC;
    const __bf16* brow1 = brow0 + (size_t)16 * CC;
    __builtin_prefetch(arow0, 0, 3);
    __builtin_prefetch(arow1, 0, 3);
    __builtin_prefetch(brow0, 0, 3);
    __builtin_prefetch(brow1, 0, 3);

    v8f c00 = {}, c01 = {}, c10 = {}, c11 = {};
    for (int k0 = 0; k0 < CC; k0 += 32) {
        v16bf a0 = load_k32(arow0 + k0, half);
        v16bf a1 = load_k32(arow1 + k0, half);
        v16bf b0 = load_k32(brow0 + k0, half);
        v16bf b1 = load_k32(brow1 + k0, half);
        c00 = wmma_bf16(a0, b0, c00);
        c01 = wmma_bf16(a0, b1, c01);
        c10 = wmma_bf16(a1, b0, c10);
        c11 = wmma_bf16(a1, b1, c11);
    }

#pragma unroll
    for (int i = 0; i < 2; ++i)
#pragma unroll
    for (int j = 0; j < 2; ++j) {
        v8f acc = (i == 0) ? (j == 0 ? c00 : c01) : (j == 0 ? c10 : c11);
        int co = cb * 32 + j * 16 + l16;
        float bias = bp[co];
#pragma unroll
        for (int r = 0; r < 8; ++r) {
            int M = r + half * 8;
            int m = rb * 32 + i * 16 + M;
            if (m < ROWS)
                out[(size_t)m * CC + co] = acc[r] + bias;
        }
    }
}

// ---------------------------------------------------------------------------
// Host side
// ---------------------------------------------------------------------------
extern "C" void kernel_launch(void* const* d_in, const int* in_sizes, int n_in,
                              void* d_out, int out_size, void* d_ws, size_t ws_size,
                              hipStream_t stream)
{
    (void)in_sizes; (void)n_in; (void)out_size; (void)ws_size;
    const float* x    = (const float*)d_in[0];
    const float* Wq   = (const float*)d_in[1];
    const float* Wk   = (const float*)d_in[2];
    const float* Wv   = (const float*)d_in[3];
    const float* Wp   = (const float*)d_in[4];
    const float* bp   = (const float*)d_in[5];
    const float* lamb = (const float*)d_in[6];
    float* out = (float*)d_out;

    // Workspace carve-up (256-byte aligned chunks)
    char* ws = (char*)d_ws;
    size_t off = 0;
    auto take = [&](size_t bytes) -> char* {
        char* p = ws + off;
        off += (bytes + 255) & ~(size_t)255;
        return p;
    };
    __bf16* xb   = (__bf16*)take((size_t)ROWS * CC * 2);          // 14.2 MB
    __bf16* wqb  = (__bf16*)take((size_t)CC * CC * 2);
    __bf16* wkb  = (__bf16*)take((size_t)CC * CC * 2);
    __bf16* wvb  = (__bf16*)take((size_t)CC * CC * 2);
    __bf16* wpb  = (__bf16*)take((size_t)CC * CC * 2);
    __bf16* qb   = (__bf16*)take((size_t)BHD * NPAD * HDIM * 2);  // 14.9 MB
    __bf16* kb   = (__bf16*)take((size_t)BHD * NPAD * HDIM * 2);
    __bf16* vtb  = (__bf16*)take((size_t)BHD * HDIM * NPAD * 2);
    __bf16* ytb  = (__bf16*)take((size_t)BHD * HDIM * NPAD * 2);
    __bf16* attn = (__bf16*)take((size_t)BHD * NPAD * NPAD * 2);  // 142 MB (L2-resident)
    __bf16* ohb  = (__bf16*)take((size_t)ROWS * CC * 2);

    // 0) fp32 -> bf16 converts
    {
        int n = ROWS * CC;
        cvt_bf16<<<(n + 255) / 256, 256, 0, stream>>>(x, xb, n);
        int m = CC * CC;
        cvt_bf16<<<(m + 255) / 256, 256, 0, stream>>>(Wq, wqb, m);
        cvt_bf16<<<(m + 255) / 256, 256, 0, stream>>>(Wk, wkb, m);
        cvt_bf16<<<(m + 255) / 256, 256, 0, stream>>>(Wv, wvb, m);
        cvt_bf16<<<(m + 255) / 256, 256, 0, stream>>>(Wp, wpb, m);
    }
    // 1) QKV projections (scale folded into Q; V transposed)
    qkv_gemm<<<dim3(NPAD / 32, BB, 3 * 24), 32, 0, stream>>>(
        xb, wqb, wkb, wvb, qb, kb, vtb);
    // 2) logits + softmax -> bf16 attn
    attn_softmax<<<dim3(NT, HH, BB), 256, 0, stream>>>(qb, kb, attn);
    // 3) Y = attn @ V  (transposed)
    av_gemm<<<dim3(NPAD / 32, 2, BHD), 32, 0, stream>>>(attn, vtb, ytb);
    // 4) Z = attn @ Y ; combine with beta(lamb) -> out_head
    az_combine<<<dim3(NPAD / 32, 2, BHD), 32, 0, stream>>>(attn, ytb, lamb, ohb);
    // 5) out = out_head @ Wp^T + bp
    out_proj<<<dim3((ROWS + 31) / 32, CC / 32), 32, 0, stream>>>(ohb, wpb, bp, out);
}
